// GMM_77859167142552
// MI455X (gfx1250) — matrix operand order repaired and checked by reading, not compiled, not run on hardware
//
#include <hip/hip_runtime.h>
#include <hip/hip_bf16.h>
#include <math.h>

typedef __attribute__((ext_vector_type(16))) __bf16 v16bf;
typedef __attribute__((ext_vector_type(8)))  float  v8f;

#define DDIM 128
#define KMIX 16
#define LOG_2PI 1.8378770664093453f

// ---------------------------------------------------------------------------
// CDNA5 async memory->LDS copy (ASYNCcnt-tracked), per-lane 16B.
// ---------------------------------------------------------------------------
__device__ __forceinline__ void async_ld_b128(void* lds_ptr, const void* gptr) {
    const unsigned ldsoff = (unsigned)(uintptr_t)lds_ptr;
    asm volatile("global_load_async_to_lds_b128 %0, %1, off"
                 :: "v"(ldsoff), "v"(gptr)
                 : "memory");
}
__device__ __forceinline__ void wait_async0() {
    asm volatile("s_wait_asynccnt 0x0" ::: "memory");
}

// ---------------------------------------------------------------------------
// Kernel 1: per-mixture Cholesky -> inverse -> logdet; emits:
//   Bhi/Blo : 0.5*inv(sigma) split into bf16 hi/lo, pre-swizzled into the
//             CDNA5 WMMA B-operand layout (32x16 blocks, lane-major)
//   Wlin    : inv(sigma) @ mu
//   Cc      : mu^T (0.5 inv) mu + 0.5*(D log2pi + logdet) - log(phi)
// ---------------------------------------------------------------------------
__global__ __launch_bounds__(128) void gmm_prep(
    const float* __restrict__ sigma, const float* __restrict__ mu,
    const float* __restrict__ phi,
    __bf16* __restrict__ Bhi, __bf16* __restrict__ Blo,
    float* __restrict__ Wlin, float* __restrict__ Cc)
{
    __shared__ float S[DDIM][DDIM];
    const int k = blockIdx.x;
    const int t = threadIdx.x;
    const float* sig = sigma + (size_t)k * DDIM * DDIM;

    for (int j = 0; j < DDIM; j += 4) {
        const float4 v = *(const float4*)(sig + t * DDIM + j);
        S[t][j] = v.x; S[t][j + 1] = v.y; S[t][j + 2] = v.z; S[t][j + 3] = v.w;
    }
    __syncthreads();

    for (int j = 0; j < DDIM; ++j) {                  // Cholesky (lower = L)
        if (t == j) S[j][j] = sqrtf(S[j][j]);
        __syncthreads();
        const float ljj = S[j][j];
        if (t > j) S[t][j] /= ljj;
        __syncthreads();
        if (t > j) {
            const float ltj = S[t][j];
            for (int c2 = j + 1; c2 <= t; ++c2) S[t][c2] -= ltj * S[c2][j];
        }
        __syncthreads();
    }

    float logdet = 0.f;
    if (t == 0) { for (int j = 0; j < DDIM; ++j) logdet += 2.f * logf(S[j][j]); }
    __syncthreads();
    S[t][t] = 1.0f / S[t][t];
    __syncthreads();

    {                                                 // Linv col c -> upper
        const int c = t;
        for (int i = c + 1; i < DDIM; ++i) {
            float sum = S[i][c] * S[c][c];
            for (int p = c + 1; p < i; ++p) sum += S[i][p] * S[c][p];
            S[c][i] = -sum * S[i][i];
        }
    }
    __syncthreads();

    {                                                 // invA col j = thread
        const int j = t;
        const float* muk = mu + k * DDIM;
        float wj = 0.f;
        for (int i = 0; i < DDIM; ++i) {
            const int lo = (i > j) ? i : j;
            float acc = 0.f;
            for (int p = lo; p < DDIM; ++p) {
                const float li = (p == i) ? S[i][i] : S[i][p];
                const float lj = (p == j) ? S[j][j] : S[j][p];
                acc += li * lj;
            }
            wj += acc * muk[i];
            const float ah = 0.5f * acc;
            const __bf16 h  = (__bf16)ah;
            const __bf16 l2 = (__bf16)(ah - (float)h);
            const int s    = i >> 5;
            const int kk   = i & 31;
            const int lane = (j & 15) + ((kk & 16) ? 16 : 0);
            const int e    = kk & 15;
            const int tt   = j >> 4;
            const int idx  = ((((k * 8 + tt) * 4 + s) * 32 + lane) * 16) + e;
            Bhi[idx] = h; Blo[idx] = l2;
        }
        Wlin[k * DDIM + j] = wj;
        __syncthreads();
        S[0][j] = wj * muk[j];
        __syncthreads();
        if (t == 0) {
            float cq = 0.f;
            for (int jj = 0; jj < DDIM; ++jj) cq += S[0][jj];
            Cc[k] = 0.5f * cq + 0.5f * (DDIM * LOG_2PI + logdet) - logf(phi[k]);
        }
    }
}

// ---------------------------------------------------------------------------
// Kernel 2: 256 threads = 8 waves x 16 rows (128 rows/block).
// B matrices async-staged to LDS with ping-pong double buffering: stage(k+1)
// issued before compute(k), waited only at iteration end (copy fully hidden).
// Buffer 1 aliases the X-tile region (dead after setup: A-operands + row-dot
// values are hoisted to registers, linear term precomputed).
// ---------------------------------------------------------------------------
#define XS_STRIDE 132                                  // pad: %64 != 0
#define XS_BYTES  (8 * 16 * XS_STRIDE * 4)             // 67584
#define BUF_BYTES 65536                                // hi 32K + lo 32K
#define SMEM_BYTES (XS_BYTES + BUF_BYTES)              // 133120

__global__ __launch_bounds__(256) void gmm_main(
    const float* __restrict__ X,
    const __bf16* __restrict__ Bhi, const __bf16* __restrict__ Blo,
    const float* __restrict__ Wlin, const float* __restrict__ Cc,
    float* __restrict__ out)
{
    __shared__ alignas(64) char smem[SMEM_BYTES];
    __shared__ float Ws[KMIX][DDIM];
    __shared__ float Ccs[KMIX];
    __shared__ float nll_s[8][16][KMIX];

    float*  Xs   = (float*)smem;                       // [8][16][132], setup only
    __bf16* bufA = (__bf16*)(smem + XS_BYTES);         // mixture buffer 0
    __bf16* bufB = (__bf16*)smem;                      // aliases Xs in k-loop

    const int tid  = threadIdx.x;
    const int wave = tid >> 5;
    const int lane = tid & 31;
    const int rowblock = blockIdx.x * 128;
    const int rowbase  = rowblock + wave * 16;

    // ---- async stage: X tile (64KB padded) + mixture-0 B into bufA ----
    #pragma unroll
    for (int i = 0; i < 16; ++i) {                     // 4096 16B chunks of X
        const int c  = tid + i * 256;
        const int r  = c >> 5;
        const int cc = c & 31;
        async_ld_b128(Xs + ((r >> 4) * 16 + (r & 15)) * XS_STRIDE + cc * 4,
                      X + (size_t)(rowblock + r) * DDIM + cc * 4);
    }
    #pragma unroll
    for (int i = 0; i < 8; ++i) {                      // 2048 chunks hi + lo
        const int c = tid + i * 256;
        async_ld_b128((char*)bufA + c * 16,         (const char*)Bhi + c * 16);
        async_ld_b128((char*)bufA + 32768 + c * 16, (const char*)Blo + c * 16);
    }
    for (int i = tid; i < KMIX * DDIM; i += 256) Ws[i >> 7][i & 127] = Wlin[i];
    if (tid < KMIX) Ccs[tid] = Cc[tid];
    wait_async0();
    __syncthreads();

    // ---- build WMMA A-operands (hi/lo) for 4 k-steps ----
    v16bf ahi[4], alo[4];
    const int m    = lane & 15;
    const int koff = (lane < 16) ? 0 : 16;
    #pragma unroll
    for (int s = 0; s < 4; ++s) {
        #pragma unroll
        for (int e = 0; e < 16; ++e) {
            const int kk = 32 * s + koff + e;
            const float v = Xs[(wave * 16 + m) * XS_STRIDE + kk];
            const __bf16 h = (__bf16)v;
            ahi[s][e] = h;
            alo[s][e] = (__bf16)(v - (float)h);
        }
    }

    // ---- hoist row-dot X values into registers: Xreg[t][r] = X[mo+r][16t+n0]
    const int n0 = lane & 15;
    const int mo = (lane < 16) ? 0 : 8;
    float Xreg[8][8];
    #pragma unroll
    for (int t = 0; t < 8; ++t)
        #pragma unroll
        for (int r = 0; r < 8; ++r)
            Xreg[t][r] = Xs[(wave * 16 + mo + r) * XS_STRIDE + t * 16 + n0];

    // ---- linear + constant:  nll_s = c_k - w_k · x ----
    {
        const int row = lane & 15;
        const int j0  = (lane < 16) ? 0 : 64;
        for (int k = 0; k < KMIX; ++k) {
            float p = 0.f;
            #pragma unroll 8
            for (int j = 0; j < 64; ++j)
                p += Ws[k][j0 + j] * Xs[(wave * 16 + row) * XS_STRIDE + j0 + j];
            p += __shfl_xor(p, 16, 32);
            if (lane < 16) nll_s[wave][row][k] = Ccs[k] - p;
        }
    }
    __syncthreads();                                   // Xs reads all done

    // ---- quadratic term: WMMA bf16x3, double-buffered LDS B ----
    for (int k = 0; k < KMIX; ++k) {
        const __bf16* buf = (k & 1) ? bufB : bufA;
        if (k < KMIX - 1) {                            // prefetch next mixture
            __bf16* nbuf = (k & 1) ? bufA : bufB;
            const char* gh = (const char*)Bhi + (size_t)(k + 1) * 32768;
            const char* gl = (const char*)Blo + (size_t)(k + 1) * 32768;
            #pragma unroll
            for (int i = 0; i < 8; ++i) {
                const int c = tid + i * 256;
                async_ld_b128((char*)nbuf + c * 16,         gh + c * 16);
                async_ld_b128((char*)nbuf + 32768 + c * 16, gl + c * 16);
            }
        }
        float qacc[8] = {0.f, 0.f, 0.f, 0.f, 0.f, 0.f, 0.f, 0.f};
        #pragma unroll
        for (int t = 0; t < 8; ++t) {                  // n-tiles
            v8f C = {0.f, 0.f, 0.f, 0.f, 0.f, 0.f, 0.f, 0.f};
            const __bf16* bh = buf + (t * 4) * 512 + lane * 16;
            const __bf16* bl = bh + 16384;
            #pragma unroll
            for (int s = 0; s < 4; ++s) {              // k-steps
                const v16bf B1 = *(const v16bf*)(bh + s * 512);
                const v16bf B2 = *(const v16bf*)(bl + s * 512);
                C = __builtin_amdgcn_wmma_f32_16x16x32_bf16(false, ahi[s], false, B1,
                                                            (short)0, C, false, false);
                C = __builtin_amdgcn_wmma_f32_16x16x32_bf16(false, ahi[s], false, B2,
                                                            (short)0, C, false, false);
                C = __builtin_amdgcn_wmma_f32_16x16x32_bf16(false, alo[s], false, B1,
                                                            (short)0, C, false, false);
            }
            #pragma unroll
            for (int r = 0; r < 8; ++r) qacc[r] += C[r] * Xreg[t][r];
        }
        #pragma unroll
        for (int r = 0; r < 8; ++r) {                  // reduce 16 cols/half
            float q = qacc[r];
            q += __shfl_xor(q, 1, 32);
            q += __shfl_xor(q, 2, 32);
            q += __shfl_xor(q, 4, 32);
            q += __shfl_xor(q, 8, 32);
            qacc[r] = q;
        }
        if (n0 == 0) {                                 // lanes 0 / 16 commit
            #pragma unroll
            for (int r = 0; r < 8; ++r) nll_s[wave][mo + r][k] += qacc[r];
        }
        if (k < KMIX - 1) wait_async0();               // copy hid behind wmma
        __syncthreads();
    }

    // ---- row-max normalize and store ----
    if (lane < 16) {
        const int row = lane;
        float mx = nll_s[wave][row][0];
        #pragma unroll
        for (int k = 1; k < KMIX; ++k) mx = fmaxf(mx, nll_s[wave][row][k]);
        const float inv = 1.0f / mx;
        #pragma unroll
        for (int k4 = 0; k4 < 4; ++k4) {
            float4 o;
            o.x = nll_s[wave][row][k4 * 4 + 0] * inv;
            o.y = nll_s[wave][row][k4 * 4 + 1] * inv;
            o.z = nll_s[wave][row][k4 * 4 + 2] * inv;
            o.w = nll_s[wave][row][k4 * 4 + 3] * inv;
            *(float4*)(out + (size_t)(rowbase + row) * KMIX + k4 * 4) = o;
        }
    }
}

// ---------------------------------------------------------------------------
extern "C" void kernel_launch(void* const* d_in, const int* in_sizes, int n_in,
                              void* d_out, int out_size, void* d_ws, size_t ws_size,
                              hipStream_t stream) {
    const float* X     = (const float*)d_in[0];   // [131072, 128]
    const float* mu    = (const float*)d_in[1];   // [16, 128]
    const float* sigma = (const float*)d_in[2];   // [16, 128, 128]
    const float* phi   = (const float*)d_in[3];   // [16]

    char* ws = (char*)d_ws;
    __bf16* Bhi = (__bf16*)ws;                          // 512 KB
    __bf16* Blo = (__bf16*)(ws + 524288);               // 512 KB
    float*  Wlin = (float*)(ws + 1048576);              // 8 KB
    float*  Cc   = (float*)(ws + 1048576 + 8192);       // 64 B

    gmm_prep<<<KMIX, 128, 0, stream>>>(sigma, mu, phi, Bhi, Blo, Wlin, Cc);

    const int N = in_sizes[0] / DDIM;                   // 131072
    gmm_main<<<N / 128, 256, 0, stream>>>(X, Bhi, Blo, Wlin, Cc, (float*)d_out);
}
